// EdgeNetEMD_60387240181866
// MI455X (gfx1250) — matrix operand order repaired and checked by reading, not compiled, not run on hardware
//
#include <hip/hip_runtime.h>
#include <stdint.h>

// ---------------------------------------------------------------------------
// EdgeConv autoencoder on MI455X (gfx1250, wave32, WMMA).
// Pipeline: BN -> EdgeConv(8->32->32->2, relu) mean-agg -> EdgeConv(4->32->32->4) mean-agg
// ---------------------------------------------------------------------------

typedef __attribute__((ext_vector_type(16))) _Float16 v16h;
typedef __attribute__((ext_vector_type(8)))  float    v8f;
typedef __attribute__((ext_vector_type(2)))  _Float16 h2;

union F16Frag { v16h v; uint32_t u[8]; _Float16 h[16]; };
union F32Frag { v8f  v; float f[8]; };
union U32H2   { uint32_t u; h2 h; };

static __device__ inline uint32_t pack2(float a, float b) {
  U32H2 t; t.h.x = (_Float16)a; t.h.y = (_Float16)b; return t.u;
}
static __device__ inline v8f splat8(float x) {
  v8f r = {x, x, x, x, x, x, x, x}; return r;
}
// One packed f16 pair of B-matrix (32x16, f16) for VGPR v of this lane.
// B layout: lanes 0-15 hold K=0..7 (V0-3) & 16..23 (V4-7); lanes 16-31 K=8..15 & 24..31.
static __device__ inline uint32_t weight_pair(const float* __restrict__ W, int ldn,
                                              int nvalid, int klim, int k0, int n) {
  float a = (k0     < klim && n < nvalid) ? W[k0 * ldn + n]       : 0.f;
  float b = (k0 + 1 < klim && n < nvalid) ? W[(k0 + 1) * ldn + n] : 0.f;
  return pack2(a, b);
}
static __device__ inline v8f wmma_f16(const F16Frag& A, const F16Frag& B, v8f C) {
  return __builtin_amdgcn_wmma_f32_16x16x32_f16(false, A.v, false, B.v,
                                                (short)0, C, false, false);
}
static __device__ inline void wave_fence() {
  __builtin_amdgcn_fence(__ATOMIC_ACQ_REL, "wavefront");
  __builtin_amdgcn_wave_barrier();
}

// ---------------------------------------------------------------------------
// Small helper kernels
// ---------------------------------------------------------------------------
__global__ void zero_kernel(float* __restrict__ p, int n) {
  for (int i = blockIdx.x * blockDim.x + threadIdx.x; i < n; i += gridDim.x * blockDim.x)
    p[i] = 0.f;
}

__global__ void bn_stats_kernel(const float* __restrict__ x, float* __restrict__ stats, int N) {
  float s[4] = {0.f, 0.f, 0.f, 0.f};
  float q[4] = {0.f, 0.f, 0.f, 0.f};
  for (int i = blockIdx.x * blockDim.x + threadIdx.x; i < N; i += gridDim.x * blockDim.x) {
#pragma unroll
    for (int c = 0; c < 4; ++c) {
      float v = x[i * 4 + c];
      s[c] += v; q[c] += v * v;
    }
  }
#pragma unroll
  for (int off = 16; off > 0; off >>= 1) {
#pragma unroll
    for (int c = 0; c < 4; ++c) {
      s[c] += __shfl_down(s[c], off, 32);
      q[c] += __shfl_down(q[c], off, 32);
    }
  }
  if ((threadIdx.x & 31) == 0) {
#pragma unroll
    for (int c = 0; c < 4; ++c) {
      atomicAdd(&stats[c], s[c]);
      atomicAdd(&stats[4 + c], q[c]);
    }
  }
}

// Normalize and pack node features into f16 pairs (2 dwords per node).
__global__ void bn_apply_kernel(const float* __restrict__ x, const float* __restrict__ stats,
                                const float* __restrict__ bnw, const float* __restrict__ bnb,
                                uint32_t* __restrict__ xp, int N) {
  int i = blockIdx.x * blockDim.x + threadIdx.x;
  if (i >= N) return;
  float o[4];
#pragma unroll
  for (int c = 0; c < 4; ++c) {
    float mu   = stats[c] / (float)N;
    float var  = stats[4 + c] / (float)N - mu * mu;
    float rstd = rsqrtf(var + 1e-5f);
    o[c] = (x[i * 4 + c] - mu) * rstd * bnw[c] + bnb[c];
  }
  xp[i * 2 + 0] = pack2(o[0], o[1]);
  xp[i * 2 + 1] = pack2(o[2], o[3]);
}

__global__ void degree_kernel(const int* __restrict__ dst, float* __restrict__ deg, int E) {
  for (int e = blockIdx.x * blockDim.x + threadIdx.x; e < E; e += gridDim.x * blockDim.x)
    atomicAdd(&deg[dst[e]], 1.f);
}

__global__ void finalize_enc_kernel(const float* __restrict__ agg1, const float* __restrict__ deg,
                                    uint32_t* __restrict__ h1p, int N) {
  int i = blockIdx.x * blockDim.x + threadIdx.x;
  if (i >= N) return;
  float inv = 1.f / fmaxf(deg[i], 1.f);
  h1p[i] = pack2(agg1[2 * i] * inv, agg1[2 * i + 1] * inv);
}

__global__ void finalize_dec_kernel(const float* __restrict__ agg2, const float* __restrict__ deg,
                                    float* __restrict__ out, int N) {
  int i = blockIdx.x * blockDim.x + threadIdx.x;
  if (i >= N) return;
  float inv = 1.f / fmaxf(deg[i], 1.f);
#pragma unroll
  for (int c = 0; c < 4; ++c) out[4 * i + c] = agg2[4 * i + c] * inv;
}

// ---------------------------------------------------------------------------
// WMMA EdgeConv kernel: one wave owns a tile of 16 edges.
//   IN_WORDS: packed-f16 dwords per node (2 -> F=4 encoder, 1 -> F=2 decoder)
//   MLP: [2F padded to 32] x [32x32 W1pad] -> relu -> [32x32 W2] -> relu
//        -> [32x16 W3pad] (OUT_DIM valid cols) (+ optional final relu)
// ---------------------------------------------------------------------------
template <int IN_WORDS, int OUT_DIM, bool FINAL_RELU>
__global__ __launch_bounds__(256) void edgeconv_kernel(
    const uint32_t* __restrict__ xp,
    const int* __restrict__ src, const int* __restrict__ dst,
    const float* __restrict__ W1, const float* __restrict__ B1,
    const float* __restrict__ W2, const float* __restrict__ B2,
    const float* __restrict__ W3, const float* __restrict__ B3,
    float* __restrict__ agg, int E) {
  constexpr int KLIM = IN_WORDS * 4;  // real input dim (2F), rest zero-padded

  struct WaveLds {
    _Float16 stage[16 * 32];  // f16 activations between layers (row-major 16x32)
    float    out[16 * 4];     // final per-edge outputs
    int      dstIdx[16];      // gather of dst indices for this tile
  };
  __shared__ WaveLds lds[8];  // blockDim.x == 256 -> 8 waves

  const int lane  = threadIdx.x & 31;
  const int w     = threadIdx.x >> 5;
  const int col   = lane & 15;
  const int khalf = lane >> 4;
  WaveLds& L = lds[w];

  // ---- loop-invariant B fragments (weights, zero-padded) and bias C frags ----
  F16Frag b1lo, b1hi, b2lo, b2hi, b3f;
#pragma unroll
  for (int v = 0; v < 8; ++v) {
    const int k0 = (v < 4) ? (khalf * 8 + 2 * v) : (16 + khalf * 8 + 2 * (v - 4));
    b1lo.u[v] = weight_pair(W1, 32, 32, KLIM, k0, col);
    b1hi.u[v] = weight_pair(W1, 32, 32, KLIM, k0, col + 16);
    b2lo.u[v] = weight_pair(W2, 32, 32, 32, k0, col);
    b2hi.u[v] = weight_pair(W2, 32, 32, 32, k0, col + 16);
    b3f.u[v]  = weight_pair(W3, OUT_DIM, OUT_DIM, 32, k0, col);
  }
  const v8f cb1lo = splat8(B1[col]);
  const v8f cb1hi = splat8(B1[col + 16]);
  const v8f cb2lo = splat8(B2[col]);
  const v8f cb2hi = splat8(B2[col + 16]);
  const v8f cb3   = splat8(col < OUT_DIM ? B3[col] : 0.f);

  const int wavesPerBlock = blockDim.x >> 5;
  const int waveId = blockIdx.x * wavesPerBlock + w;
  const int nWaves = gridDim.x * wavesPerBlock;
  const int tiles  = (E + 15) >> 4;

  for (int t = waveId; t < tiles; t += nWaves) {
    const int eBase = t << 4;

    // ---- build A fragment for layer 1 directly in registers ----
    // 16-bit A 16x32 layout: lanes 0-15 row M=lane hold K=0..7 (V0-3) & 16..23 (V4-7);
    // lanes 16-31 hold K=8..15 & 24..31 -> all zero here (input dim <= 8).
    F16Frag a;
#pragma unroll
    for (int v = 0; v < 8; ++v) a.u[v] = 0u;
    if (lane < 16) {
      int e = eBase + lane;
      if (e >= E) e = E - 1;  // clamp; masked at aggregation
      const int s = src[e];
      const int d = dst[e];
      L.dstIdx[lane] = d;
#pragma unroll
      for (int q = 0; q < IN_WORDS; ++q) {
        U32H2 xi, xj, df;
        xi.u = xp[d * IN_WORDS + q];
        xj.u = xp[s * IN_WORDS + q];
        df.h = xj.h - xi.h;          // x_j - x_i in f16
        a.u[q]            = xi.u;    // K = [0, F)    : x_i
        a.u[IN_WORDS + q] = df.u;    // K = [2F-? ..) : x_j - x_i
      }
    }
    wave_fence();

    // ---- layer 1: [16x32] x [32x32] + b1, relu ----
    F32Frag d0, d1;
    d0.v = wmma_f16(a, b1lo, cb1lo);
    d1.v = wmma_f16(a, b1hi, cb1hi);
#pragma unroll
    for (int v = 0; v < 8; ++v) {
      const int m = khalf * 8 + v;  // D layout: lane=col, VGPR v -> row m
      float x0 = d0.f[v]; x0 = x0 > 0.f ? x0 : 0.f;
      float x1 = d1.f[v]; x1 = x1 > 0.f ? x1 : 0.f;
      L.stage[m * 32 + col]      = (_Float16)x0;
      L.stage[m * 32 + col + 16] = (_Float16)x1;
    }
    wave_fence();

    // ---- reload as A fragment for layer 2 ----
    F16Frag a2;
    {
      const uint32_t* s32 = (const uint32_t*)L.stage;
#pragma unroll
      for (int v = 0; v < 8; ++v) {
        const int k = (v < 4) ? (khalf * 8 + 2 * v) : (16 + khalf * 8 + 2 * (v - 4));
        a2.u[v] = s32[(col * 32 + k) >> 1];
      }
    }
    wave_fence();

    // ---- layer 2: [16x32] x [32x32] + b2, relu ----
    d0.v = wmma_f16(a2, b2lo, cb2lo);
    d1.v = wmma_f16(a2, b2hi, cb2hi);
#pragma unroll
    for (int v = 0; v < 8; ++v) {
      const int m = khalf * 8 + v;
      float x0 = d0.f[v]; x0 = x0 > 0.f ? x0 : 0.f;
      float x1 = d1.f[v]; x1 = x1 > 0.f ? x1 : 0.f;
      L.stage[m * 32 + col]      = (_Float16)x0;
      L.stage[m * 32 + col + 16] = (_Float16)x1;
    }
    wave_fence();

    F16Frag a3;
    {
      const uint32_t* s32 = (const uint32_t*)L.stage;
#pragma unroll
      for (int v = 0; v < 8; ++v) {
        const int k = (v < 4) ? (khalf * 8 + 2 * v) : (16 + khalf * 8 + 2 * (v - 4));
        a3.u[v] = s32[(col * 32 + k) >> 1];
      }
    }
    wave_fence();

    // ---- layer 3: [16x32] x [32xOUT_DIM padded to 16] + b3 ----
    F32Frag o;
    o.v = wmma_f16(a3, b3f, cb3);
    if (col < OUT_DIM) {
#pragma unroll
      for (int v = 0; v < 8; ++v) L.out[(khalf * 8 + v) * OUT_DIM + col] = o.f[v];
    }
    wave_fence();

    // ---- scatter-add messages into agg[dst] (mean computed in finalize) ----
    for (int i = lane; i < 16 * OUT_DIM; i += 32) {
      const int r = i / OUT_DIM;
      const int c = i % OUT_DIM;
      if (eBase + r < E) {
        float vv = L.out[r * OUT_DIM + c];
        if (FINAL_RELU) vv = vv > 0.f ? vv : 0.f;
        atomicAdd(&agg[L.dstIdx[r] * OUT_DIM + c], vv);
      }
    }
    wave_fence();
  }
}

// ---------------------------------------------------------------------------
// Host launcher
// ---------------------------------------------------------------------------
extern "C" void kernel_launch(void* const* d_in, const int* in_sizes, int n_in,
                              void* d_out, int out_size, void* d_ws, size_t ws_size,
                              hipStream_t stream) {
  const float* x    = (const float*)d_in[0];
  const int*   ei   = (const int*)d_in[1];
  const float* bn_w = (const float*)d_in[2];
  const float* bn_b = (const float*)d_in[3];
  const float* ew1  = (const float*)d_in[4];
  const float* eb1  = (const float*)d_in[5];
  const float* ew2  = (const float*)d_in[6];
  const float* eb2  = (const float*)d_in[7];
  const float* ew3  = (const float*)d_in[8];
  const float* eb3  = (const float*)d_in[9];
  const float* dw1  = (const float*)d_in[10];
  const float* db1  = (const float*)d_in[11];
  const float* dw2  = (const float*)d_in[12];
  const float* db2  = (const float*)d_in[13];
  const float* dw3  = (const float*)d_in[14];
  const float* db3  = (const float*)d_in[15];

  const int N = in_sizes[0] / 4;   // 100000 nodes
  const int E = in_sizes[1] / 2;   // 3200000 edges
  const int* src = ei;
  const int* dst = ei + E;

  // Workspace layout (bytes): stats | deg | agg1 | agg2 | xp | h1p
  char* ws = (char*)d_ws;
  float*    stats = (float*)ws;                                    // 16 floats
  float*    deg   = (float*)(ws + 64);                             // N floats
  float*    agg1  = (float*)(ws + 64 + 4L * N);                    // 2N floats
  float*    agg2  = (float*)(ws + 64 + 12L * N);                   // 4N floats
  uint32_t* xp    = (uint32_t*)(ws + 64 + 28L * N);                // 2N dwords
  uint32_t* h1p   = (uint32_t*)(ws + 64 + 36L * N);                // N dwords

  // Zero stats + deg + agg1 + agg2 (contiguous: 16 + 7N floats)
  zero_kernel<<<2048, 256, 0, stream>>>(stats, 16 + 7 * N);

  // BatchNorm (training-mode batch stats)
  bn_stats_kernel<<<512, 256, 0, stream>>>(x, stats, N);
  bn_apply_kernel<<<(N + 255) / 256, 256, 0, stream>>>(x, stats, bn_w, bn_b, xp, N);

  // In-degree (shared by both convs: same dst)
  degree_kernel<<<2048, 256, 0, stream>>>(dst, deg, E);

  // Encoder EdgeConv: F=4 (IN_WORDS=2), out 2, final relu
  edgeconv_kernel<2, 2, true><<<2048, 256, 0, stream>>>(
      xp, src, dst, ew1, eb1, ew2, eb2, ew3, eb3, agg1, E);
  finalize_enc_kernel<<<(N + 255) / 256, 256, 0, stream>>>(agg1, deg, h1p, N);

  // Decoder EdgeConv: F=2 (IN_WORDS=1), out 4, no final relu
  edgeconv_kernel<1, 4, false><<<2048, 256, 0, stream>>>(
      h1p, src, dst, dw1, db1, dw2, db2, dw3, db3, agg2, E);
  finalize_dec_kernel<<<(N + 255) / 256, 256, 0, stream>>>(agg2, deg, (float*)d_out, N);
}
